// SizeConditionedDepthExperts_49426483642688
// MI455X (gfx1250) — compile-verified, matrix operand order checked by reading, not run on hardware
//
#include <hip/hip_runtime.h>
#include <stdint.h>

#define B_TOK 8192
#define D_IN  1024
#define NE    8
#define KCLS  5
#define H_DIM 1024

#define BM  64
#define BN  128
#define BKC 64
#define A_LD 72   // halfs per LDS row; 144 B stride = 9*16 -> b128-aligned, bank-staggered
#define B_LD 72

typedef __attribute__((ext_vector_type(16))) __bf16 v16bf;
typedef __attribute__((ext_vector_type(8)))  float  v8f;

// CDNA5 async global->LDS path (guarded; falls back to sync b128 copies)
#if defined(__has_builtin)
#if __has_builtin(__builtin_amdgcn_global_load_async_to_lds_b128) && \
    __has_builtin(__builtin_amdgcn_s_wait_asynccnt)
#define USE_ASYNC 1
#endif
#endif
#ifndef USE_ASYNC
#define USE_ASYNC 0
#endif

#if USE_ASYNC
typedef __attribute__((ext_vector_type(4))) int v4i;
typedef __attribute__((address_space(1))) v4i* as1_v4i_p;   // global int4*
typedef __attribute__((address_space(3))) v4i* as3_v4i_p;   // LDS int4*
#endif

__device__ __forceinline__ unsigned short f2bf(float f) {
  union { float f; unsigned u; } c; c.f = f;
  unsigned u = c.u;
  unsigned r = u + 0x7FFFu + ((u >> 16) & 1u);   // round-to-nearest-even
  return (unsigned short)(r >> 16);
}

__device__ __forceinline__ float bf2f(unsigned short h) {
  union { unsigned u; float f; } c; c.u = ((unsigned)h) << 16;
  return c.f;
}

// ---------------- kernel 0: zero per-expert counters ----------------
__global__ void k_zero(int* cnt) {
  if (threadIdx.x < NE) cnt[threadIdx.x] = 0;
}

// ---------------- kernel 1: build compacted per-expert token lists ----------------
__global__ void k_route(const int* __restrict__ sidx, int* cnt, int* lists) {
  int b = blockIdx.x * blockDim.x + threadIdx.x;
  if (b < B_TOK) {
    int e = sidx[b] & (NE - 1);
    int pos = atomicAdd(&cnt[e], 1);
    lists[e * B_TOK + pos] = b;
  }
}

// ---------------- kernel 2a: x (fp32) -> xb (bf16), one shot ----------------
__global__ __launch_bounds__(256)
void k_cvt_x(const float* __restrict__ x, unsigned short* __restrict__ xb) {
  size_t i = ((size_t)blockIdx.x * 256 + threadIdx.x) * 8;
  float4 a = *(const float4*)(x + i);
  float4 b = *(const float4*)(x + i + 4);
  uint4 o;
  o.x = (unsigned)f2bf(a.x) | ((unsigned)f2bf(a.y) << 16);
  o.y = (unsigned)f2bf(a.z) | ((unsigned)f2bf(a.w) << 16);
  o.z = (unsigned)f2bf(b.x) | ((unsigned)f2bf(b.y) << 16);
  o.w = (unsigned)f2bf(b.z) | ((unsigned)f2bf(b.w) << 16);
  *(uint4*)(xb + i) = o;
}

// ---------------- kernel 2b: W1 [E][D][H] fp32 -> w1t [E][H][D] bf16 ----------------
// LDS-tiled 32x32 transpose so both global sides are coalesced.
__global__ __launch_bounds__(256)
void k_w1_transpose(const float* __restrict__ W1, unsigned short* __restrict__ w1t) {
  __shared__ float tile[32][33];
  const int e  = blockIdx.z;
  const int d0 = blockIdx.x * 32;
  const int h0 = blockIdx.y * 32;
  const int t  = threadIdx.x;
  const int r  = t >> 3;          // 0..31
  const int c4 = (t & 7) << 2;    // 0,4,..,28

  float4 v = *(const float4*)(W1 + ((size_t)e * D_IN + d0 + r) * H_DIM + h0 + c4);
  tile[r][c4 + 0] = v.x; tile[r][c4 + 1] = v.y;
  tile[r][c4 + 2] = v.z; tile[r][c4 + 3] = v.w;
  __syncthreads();

  // out row = h (n-major), out col = d (k-contiguous)
  unsigned lo = (unsigned)f2bf(tile[c4 + 0][r]) | ((unsigned)f2bf(tile[c4 + 1][r]) << 16);
  unsigned hi = (unsigned)f2bf(tile[c4 + 2][r]) | ((unsigned)f2bf(tile[c4 + 3][r]) << 16);
  *(uint2*)(w1t + ((size_t)e * H_DIM + h0 + r) * D_IN + d0 + c4) = make_uint2(lo, hi);
}

// ---------------- kernel 3: per-expert layer-1 GEMM via bf16 WMMA ----------------
// grid: (H/BN, B/BM, E), block: 256 threads (8 wave32)
__global__ __launch_bounds__(256)
void k_expert_gemm(const unsigned short* __restrict__ xb,
                   const unsigned short* __restrict__ w1t,
                   const float* __restrict__ b1,
                   const int*   __restrict__ cnt,
                   const int*   __restrict__ lists,
                   unsigned short* __restrict__ h_ws) {
  __shared__ unsigned short As[BM * A_LD];  // [row][k] bf16, k-contiguous
  __shared__ unsigned short Bs[BN * B_LD];  // [n][k]  bf16, k-contiguous
  __shared__ int tok[BM];

  const int e  = blockIdx.z;
  const int m0 = blockIdx.y * BM;
  const int n0 = blockIdx.x * BN;
  const int nt = cnt[e];
  if (m0 >= nt) return;

  const int t = threadIdx.x;
  if (t < BM) {
    int g = m0 + t;
    tok[t] = (g < nt) ? lists[e * B_TOK + g] : -1;
  }
  __syncthreads();

  const int lane  = t & 31;
  const int wid   = t >> 5;
  const int wm    = wid & 3;              // m-subtile (16 rows)
  const int wn    = wid >> 2;             // n-half (64 cols)
  const int r16   = lane & 15;
  const int kg    = (lane >> 4) << 3;     // 0 or 8 (16-bit A/B lane K-group)
  const int mhalf = (lane >> 4) << 3;     // C/D layout: lanes 16-31 hold M+8

  v8f acc[4] = {v8f{}, v8f{}, v8f{}, v8f{}};

  const unsigned short* w1e = w1t + (size_t)e * H_DIM * D_IN;

  for (int kc = 0; kc < D_IN; kc += BKC) {
    // ---- stage A: 64 rows x 64 halfs (8 KB) as 512 x 16B segments ----
#pragma unroll
    for (int i = 0; i < 2; ++i) {
      int seg = t * 2 + i;
      int row = seg >> 3;
      int s8  = (seg & 7) << 3;
      int tk  = tok[row]; tk = tk < 0 ? 0 : tk;
      const unsigned short* gp = xb + (size_t)tk * D_IN + kc + s8;
      unsigned short* lp = &As[row * A_LD + s8];
#if USE_ASYNC
      __builtin_amdgcn_global_load_async_to_lds_b128(
          (as1_v4i_p)gp, (as3_v4i_p)lp, 0, 0);
#else
      *(uint4*)lp = *(const uint4*)gp;
#endif
    }
    // ---- stage B: 128 n-rows x 64 halfs (16 KB) as 1024 x 16B segments ----
#pragma unroll
    for (int i = 0; i < 4; ++i) {
      int seg = t * 4 + i;
      int n   = seg >> 3;
      int s8  = (seg & 7) << 3;
      const unsigned short* gp = w1e + (size_t)(n0 + n) * D_IN + kc + s8;
      if (kc + BKC < D_IN) __builtin_prefetch(gp + BKC, 0, 1);
      unsigned short* lp = &Bs[n * B_LD + s8];
#if USE_ASYNC
      __builtin_amdgcn_global_load_async_to_lds_b128(
          (as1_v4i_p)gp, (as3_v4i_p)lp, 0, 0);
#else
      *(uint4*)lp = *(const uint4*)gp;
#endif
    }
#if USE_ASYNC
    __builtin_amdgcn_s_wait_asynccnt(0);
#endif
    __syncthreads();

    // ---- compute: 2 k-steps x 4 n-subtiles = 8 WMMA per chunk ----
#pragma unroll
    for (int ks = 0; ks < BKC; ks += 32) {
      union { unsigned u[8]; v16bf v; } af;
      const unsigned short* ap = &As[(wm * 16 + r16) * A_LD + ks + kg];
#pragma unroll
      for (int i = 0; i < 4; ++i) {
        af.u[i]     = *(const unsigned*)(ap + 2 * i);       // K = ks+kg+{0..7}
        af.u[i + 4] = *(const unsigned*)(ap + 16 + 2 * i);  // K = ks+kg+16+{0..7}
      }
#pragma unroll
      for (int j = 0; j < 4; ++j) {
        union { unsigned u[8]; v16bf v; } bfv;
        const unsigned short* bp = &Bs[(wn * 64 + j * 16 + r16) * B_LD + ks + kg];
#pragma unroll
        for (int i = 0; i < 4; ++i) {
          bfv.u[i]     = *(const unsigned*)(bp + 2 * i);
          bfv.u[i + 4] = *(const unsigned*)(bp + 16 + 2 * i);
        }
        acc[j] = __builtin_amdgcn_wmma_f32_16x16x32_bf16(
            false, af.v, false, bfv.v, (short)0, acc[j], false, false);
      }
    }
    __syncthreads();
  }

  // ---- epilogue: +b1, ReLU, scatter bf16 h rows by original token id ----
#pragma unroll
  for (int j = 0; j < 4; ++j) {
    int gn = n0 + wn * 64 + j * 16 + r16;
    float bb = b1[e * H_DIM + gn];
#pragma unroll
    for (int r = 0; r < 8; ++r) {
      int m  = r + mhalf;
      int tk = tok[wm * 16 + m];
      if (tk >= 0) {
        float v = acc[j][r] + bb;
        v = v > 0.f ? v : 0.f;
        h_ws[(size_t)tk * H_DIM + gn] = f2bf(v);
      }
    }
  }
}

// ---------------- kernel 4: layer 2 (H x 4) + ordinal probs ----------------
// one wave32 per token; 4 tokens per 128-thread block
__global__ __launch_bounds__(128)
void k_head(const unsigned short* __restrict__ h_ws,
            const int*   __restrict__ sidx,
            const float* __restrict__ W2,
            const float* __restrict__ b2,
            float* __restrict__ out) {
  const int lane = threadIdx.x & 31;
  const int w    = threadIdx.x >> 5;
  const int b    = blockIdx.x * 4 + w;
  if (b >= B_TOK) return;

  const int e = sidx[b] & (NE - 1);
  const float* W2e = W2 + (size_t)e * H_DIM * (KCLS - 1);
  const unsigned short* hrow = h_ws + (size_t)b * H_DIM;

  float a0 = 0.f, a1 = 0.f, a2 = 0.f, a3 = 0.f;
  for (int k = lane; k < H_DIM; k += 32) {
    float hv = bf2f(hrow[k]);
    float4 wv = *(const float4*)(W2e + k * 4);
    a0 += hv * wv.x; a1 += hv * wv.y; a2 += hv * wv.z; a3 += hv * wv.w;
  }
#pragma unroll
  for (int off = 16; off > 0; off >>= 1) {
    a0 += __shfl_xor(a0, off, 32);
    a1 += __shfl_xor(a1, off, 32);
    a2 += __shfl_xor(a2, off, 32);
    a3 += __shfl_xor(a3, off, 32);
  }
  if (lane == 0) {
    float l0 = a0 + b2[e * 4 + 0];
    float l1 = a1 + b2[e * 4 + 1];
    float l2 = a2 + b2[e * 4 + 2];
    float l3 = a3 + b2[e * 4 + 3];
    out[b * 4 + 0] = l0; out[b * 4 + 1] = l1;
    out[b * 4 + 2] = l2; out[b * 4 + 3] = l3;

    float q0 = 1.f / (1.f + __expf(-l0));
    float q1 = 1.f / (1.f + __expf(-l1));
    float q2 = 1.f / (1.f + __expf(-l2));
    float q3 = 1.f / (1.f + __expf(-l3));
    float p0 = fmaxf(1.f - q0, 1e-8f);
    float p1 = fmaxf(q0 - q1, 1e-8f);
    float p2 = fmaxf(q1 - q2, 1e-8f);
    float p3 = fmaxf(q2 - q3, 1e-8f);
    float p4 = fmaxf(q3,       1e-8f);
    float inv = 1.f / fmaxf(p0 + p1 + p2 + p3 + p4, 1e-8f);
    float* pr = out + (size_t)B_TOK * 4 + (size_t)b * 5;
    pr[0] = p0 * inv; pr[1] = p1 * inv; pr[2] = p2 * inv;
    pr[3] = p3 * inv; pr[4] = p4 * inv;
  }
}

extern "C" void kernel_launch(void* const* d_in, const int* in_sizes, int n_in,
                              void* d_out, int out_size, void* d_ws, size_t ws_size,
                              hipStream_t stream) {
  const float* x    = (const float*)d_in[0];
  const int*   sidx = (const int*)d_in[1];
  const float* W1   = (const float*)d_in[2];
  const float* b1   = (const float*)d_in[3];
  const float* W2   = (const float*)d_in[4];
  const float* b2   = (const float*)d_in[5];
  float* out = (float*)d_out;

  char* ws = (char*)d_ws;
  size_t off = 0;
  int* cnt   = (int*)(ws + off); off += 256;                       // NE ints
  int* lists = (int*)(ws + off); off += (size_t)NE * B_TOK * 4;    // 256 KB
  unsigned short* h_ws = (unsigned short*)(ws + off);
  off += (size_t)B_TOK * H_DIM * 2;                                // 16 MB
  unsigned short* xb   = (unsigned short*)(ws + off);
  off += (size_t)B_TOK * D_IN * 2;                                 // 16 MB
  unsigned short* w1t  = (unsigned short*)(ws + off);              // 16 MB

  k_zero<<<1, 32, 0, stream>>>(cnt);
  k_route<<<B_TOK / 256, 256, 0, stream>>>(sidx, cnt, lists);
  k_cvt_x<<<(B_TOK * D_IN) / (256 * 8), 256, 0, stream>>>(x, xb);
  dim3 gt(D_IN / 32, H_DIM / 32, NE);
  k_w1_transpose<<<gt, 256, 0, stream>>>(W1, w1t);
  dim3 g2(H_DIM / BN, B_TOK / BM, NE);
  k_expert_gemm<<<g2, 256, 0, stream>>>(xb, w1t, b1, cnt, lists, h_ws);
  k_head<<<B_TOK / 4, 128, 0, stream>>>(h_ws, sidx, W2, b2, out);
}
